// GPT2PartialSharedAttention_55181739819323
// MI455X (gfx1250) — matrix operand order repaired and checked by reading, not compiled, not run on hardware
//
#include <hip/hip_runtime.h>
#include <math.h>

// ---------------------------------------------------------------------------
// Types for WMMA fragments (CDNA5 / gfx1250, wave32)
// ---------------------------------------------------------------------------
typedef __attribute__((ext_vector_type(16))) __bf16 v16bf;
typedef __attribute__((ext_vector_type(8)))  float  v8f;
typedef __attribute__((ext_vector_type(4)))  unsigned int u32x4;

struct Frag16 { u32x4 lo, hi; };   // 32 bytes = 16 bf16 elements per lane

static __device__ __forceinline__ v16bf as_v16bf(const Frag16& f) {
  return __builtin_bit_cast(v16bf, f);
}

static __device__ __forceinline__ Frag16 load_frag(const __bf16* p0, const __bf16* p1) {
  Frag16 f;
  f.lo = *(const u32x4*)p0;
  f.hi = *(const u32x4*)p1;
  return f;
}

static __device__ __forceinline__ v8f wmma_bf16(const Frag16& a, const Frag16& b, v8f c) {
  // D = A(16x32 bf16) * B(32x16 bf16) + C(16x16 f32)
  return __builtin_amdgcn_wmma_f32_16x16x32_bf16(
      /*neg_a=*/false, as_v16bf(a), /*neg_b=*/false, as_v16bf(b),
      /*c_mod=*/(short)0, c, /*reuse_a=*/false, /*reuse_b=*/false);
}

// Bare v_max_num_f32 (skip fmaxf's sNaN canonicalization; WMMA outputs are
// never signaling NaNs).
static __device__ __forceinline__ float fmax_raw(float a, float b) {
  float d;
  asm("v_max_num_f32 %0, %1, %2" : "=v"(d) : "v"(a), "v"(b));
  return d;
}

// ---------------------------------------------------------------------------
// DPP16 cross-lane max over a 16-lane half: pure VALU, 2 ops per step.
// ---------------------------------------------------------------------------
template<int CTRL>
static __device__ __forceinline__ float dpp_movf(float v) {
  int i = __builtin_bit_cast(int, v);
  i = __builtin_amdgcn_update_dpp(i, i, CTRL, 0xF, 0xF, true);
  return __builtin_bit_cast(float, i);
}
static __device__ __forceinline__ float half_max(float x) {
  x = fmax_raw(x, dpp_movf<0xB1>(x));    // quad_perm xor1
  x = fmax_raw(x, dpp_movf<0x4E>(x));    // quad_perm xor2
  x = fmax_raw(x, dpp_movf<0x141>(x));   // row_half_mirror
  x = fmax_raw(x, dpp_movf<0x140>(x));   // row_mirror
  return x;
}

// ---------------------------------------------------------------------------
// Problem constants
// ---------------------------------------------------------------------------
#define BB   4
#define TT   2048
#define CC   1024
#define HH   16
#define HD   64
#define SHR  16
#define PRV  48
#define MROWS (BB*TT)            // 8192
#define QSCALE 0.18033688011112042f   // log2(e) / sqrt(64), folded into Q

// ---------------------------------------------------------------------------
// Elementwise pack kernels
// ---------------------------------------------------------------------------
__global__ void cvt_f32_bf16(const float* __restrict__ in, __bf16* __restrict__ out, int n) {
  int i = blockIdx.x * blockDim.x + threadIdx.x;
  if (i < n) out[i] = (__bf16)in[i];
}

// W is (K x N) row-major (in-major).  Wt becomes (N x K) row-major bf16.
__global__ void transpose_bf16(const float* __restrict__ W, __bf16* __restrict__ Wt,
                               int K, int N) {
  int i = blockIdx.x * blockDim.x + threadIdx.x;
  if (i >= K * N) return;
  int k = i / N, n = i % N;
  Wt[(size_t)n * K + k] = (__bf16)W[i];
}

// Build Q,K (B,H,T,64) with shared dims broadcast, and V transposed (B,H,64,T).
// Q pre-scaled by log2(e)/sqrt(head_dim): scores land in base-2 softmax domain.
__global__ void repack_qkv(const __bf16* __restrict__ sh, const __bf16* __restrict__ qp,
                           const __bf16* __restrict__ kp, const __bf16* __restrict__ vp,
                           __bf16* __restrict__ Q, __bf16* __restrict__ Kt,
                           __bf16* __restrict__ Vt) {
  int i = blockIdx.x * blockDim.x + threadIdx.x;     // over B*H*T*64
  if (i >= BB * HH * TT * HD) return;
  int d = i & 63;
  int t = (i >> 6) & (TT - 1);
  int h = (i >> 17) & (HH - 1);
  int b = i >> 21;
  size_t bt = (size_t)(b * TT + t);
  float qv, kv;
  if (d < SHR) {
    float s = (float)sh[bt * SHR + d];
    qv = s; kv = s;
  } else {
    qv = (float)qp[bt * (HH * PRV) + h * PRV + (d - SHR)];
    kv = (float)kp[bt * (HH * PRV) + h * PRV + (d - SHR)];
  }
  // i itself equals ((b*H + h)*T + t)*64 + d
  Q[i]  = (__bf16)(qv * QSCALE);
  Kt[i] = (__bf16)kv;
  Vt[(((size_t)(b * HH + h)) * HD + d) * TT + t] = vp[bt * CC + h * HD + d];
}

// ---------------------------------------------------------------------------
// WMMA GEMM:  C(M x N) = A(M x K, bf16 row-major) * Bt^T (Bt is N x K bf16)
// One wave computes a 64 x (up to 64) tile: 16 f32 accumulators, K step 32.
// Each B fragment is reused by 4 WMMAs -> 16 bytes/lane per WMMA.
// ---------------------------------------------------------------------------
template<int OUTF32>
__global__ void gemm_wmma(const __bf16* __restrict__ A, const __bf16* __restrict__ Bt,
                          const float* __restrict__ bias, void* __restrict__ Cout,
                          int M, int N, int K) {
  const int lane = threadIdx.x & 31;
  const int wid  = blockIdx.x * (blockDim.x >> 5) + (threadIdx.x >> 5);
  const int nt16 = N >> 4;
  const int ngrp = (nt16 + 3) >> 2;
  const int mt   = wid / ngrp;
  const int ng   = wid % ngrp;
  if (mt >= (M >> 6)) return;
  int jmax = nt16 - ng * 4; if (jmax > 4) jmax = 4;
  const int m0 = mt << 6;
  const int n0 = ng << 6;
  const int nl = lane & 15, half = lane >> 4;

  v8f acc[4][4] = {};
  const __bf16* arow = A + (size_t)(m0 + nl) * K + half * 8;
  for (int k0 = 0; k0 < K; k0 += 32) {
    Frag16 af[4];
#pragma unroll
    for (int mi = 0; mi < 4; ++mi) {
      const __bf16* p = arow + (size_t)(mi * 16) * K + k0;
      af[mi] = load_frag(p, p + 16);
    }
#pragma unroll
    for (int j = 0; j < 4; ++j) {
      if (j < jmax) {
        const __bf16* brow = Bt + (size_t)(n0 + j * 16 + nl) * K + k0 + half * 8;
        Frag16 bf = load_frag(brow, brow + 16);
#pragma unroll
        for (int mi = 0; mi < 4; ++mi)
          acc[mi][j] = wmma_bf16(af[mi], bf, acc[mi][j]);
      }
    }
  }
#pragma unroll
  for (int mi = 0; mi < 4; ++mi) {
#pragma unroll
    for (int j = 0; j < 4; ++j) {
      if (j >= jmax) continue;
      const int col = n0 + j * 16 + nl;        // C-layout: N index = lane%16
      const float bv = bias ? bias[col] : 0.0f;
#pragma unroll
      for (int r = 0; r < 8; ++r) {            // M index = r + 8*(lane/16)
        const int row = m0 + mi * 16 + r + 8 * half;
        const float v = acc[mi][j][r] + bv;
        if (OUTF32) ((float*)Cout)[(size_t)row * N + col] = v;
        else        ((__bf16*)Cout)[(size_t)row * N + col] = (__bf16)v;
      }
    }
  }
}

// ---------------------------------------------------------------------------
// Flash attention: one wave per (b, h, 16-query tile). Causal, streaming
// softmax in base-2.  Only the diagonal 32-key block is masked.  Row maxima
// via DPP16; row sums via an extra WMMA against a constant all-ones B
// fragment (the resulting accumulator IS the normalizer, alpha-rescaled in
// lockstep with O and already broadcast across lanes for the epilogue).
// ---------------------------------------------------------------------------
__global__ void flash_attn(const __bf16* __restrict__ Q, const __bf16* __restrict__ Kt,
                           const __bf16* __restrict__ Vt, __bf16* __restrict__ Y) {
  __shared__ __align__(16) __bf16 Plds[8][16 * 32];
  const int lane = threadIdx.x & 31;
  const int w    = threadIdx.x >> 5;
  const int wid  = blockIdx.x * 8 + w;                 // [0, 8192)
  const int qt = wid & 127;
  const int h  = (wid >> 7) & (HH - 1);
  const int b  = wid >> 11;
  const int q0 = qt << 4;
  const int nl = lane & 15, half = lane >> 4;

  const size_t bh = (size_t)(b * HH + h);
  const __bf16* qbase = Q + (bh * TT + q0) * HD;

  Frag16 qf[2];
#pragma unroll
  for (int dk = 0; dk < 2; ++dk) {
    const __bf16* p = qbase + nl * HD + dk * 32 + half * 8;
    qf[dk] = load_frag(p, p + 16);
  }

  // constant all-ones B fragment (bf16 1.0 pairs); rowsum = P @ ones
  Frag16 onesf;
  onesf.lo = (u32x4)(0x3F803F80u);
  onesf.hi = (u32x4)(0x3F803F80u);

  v8f o[4] = {};
  v8f ol = {};                     // P@1 accumulator == softmax denominator
  float mr[8];
#pragma unroll
  for (int r = 0; r < 8; ++r) mr[r] = -__builtin_inff();

  __bf16* pl = &Plds[w][0];
  const __bf16* vbase = Vt + bh * HD * TT;
  const __bf16* kroot = Kt + bh * TT * HD;

  auto process_block = [&](int kb, bool masked) {
    const __bf16* kbase = kroot + (size_t)kb * HD;
    v8f s[2] = {};
#pragma unroll
    for (int st = 0; st < 2; ++st) {
      const __bf16* krow = kbase + (st * 16 + nl) * HD;
#pragma unroll
      for (int dk = 0; dk < 2; ++dk) {
        const __bf16* p = krow + dk * 32 + half * 8;
        Frag16 kf = load_frag(p, p + 16);
        s[st] = wmma_bf16(qf[dk], kf, s[st]);
      }
    }
#pragma unroll
    for (int r = 0; r < 8; ++r) {
      float x0 = s[0][r];
      float x1 = s[1][r];
      if (masked) {
        const int qrow = q0 + r + 8 * half;
        const int key0 = kb + nl;
        x0 = (key0      <= qrow) ? x0 : -__builtin_inff();
        x1 = (key0 + 16 <= qrow) ? x1 : -__builtin_inff();
      }
      const float rm = half_max(fmax_raw(x0, x1));
      const float nm = fmax_raw(mr[r], rm);
      const float alpha = __builtin_amdgcn_exp2f(mr[r] - nm);
      const float p0 = __builtin_amdgcn_exp2f(x0 - nm);
      const float p1 = __builtin_amdgcn_exp2f(x1 - nm);
      mr[r] = nm;
      const int row = r + 8 * half;
      pl[row * 32 + nl]      = (__bf16)p0;
      pl[row * 32 + 16 + nl] = (__bf16)p1;
      o[0][r] *= alpha; o[1][r] *= alpha; o[2][r] *= alpha; o[3][r] *= alpha;
      ol[r]   *= alpha;
    }
    // wave-synchronous LDS handoff (C-layout stores -> A-layout reload)
    asm volatile("s_wait_dscnt 0x0" ::: "memory");
    Frag16 pf;
    const __bf16* pp = pl + nl * 32 + half * 8;
    pf.lo = *(const u32x4*)pp;
    pf.hi = *(const u32x4*)(pp + 16);
#pragma unroll
    for (int j = 0; j < 4; ++j) {
      const __bf16* vrow = vbase + (size_t)(j * 16 + nl) * TT + kb + half * 8;
      Frag16 vf = load_frag(vrow, vrow + 16);
      o[j] = wmma_bf16(pf, vf, o[j]);
    }
    ol = wmma_bf16(pf, onesf, ol);   // row sums, broadcast across all lanes
  };

  // steady state: fully-unmasked blocks, then one masked diagonal block
  const int kb_last = ((q0 + 15) >> 5) << 5;
  for (int kb = 0; kb < kb_last; kb += 32) process_block(kb, false);
  process_block(kb_last, true);

  // normalize and store Y in (B, T, C) order, C index = h*64 + d
#pragma unroll
  for (int r = 0; r < 8; ++r) {
    const float inv = 1.0f / ol[r];
    const int qrow = q0 + r + 8 * half;
    const size_t base = ((size_t)(b * TT + qrow)) * CC + h * HD + nl;
#pragma unroll
    for (int j = 0; j < 4; ++j) {
      Y[base + j * 16] = (__bf16)(o[j][r] * inv);
    }
  }
}

// ---------------------------------------------------------------------------
// Host-side orchestration
// ---------------------------------------------------------------------------
extern "C" void kernel_launch(void* const* d_in, const int* in_sizes, int n_in,
                              void* d_out, int out_size, void* d_ws, size_t ws_size,
                              hipStream_t stream) {
  const float* x   = (const float*)d_in[0];
  const float* Wsh = (const float*)d_in[1];
  const float* bsh = (const float*)d_in[2];
  const float* Wq  = (const float*)d_in[3];
  const float* bq  = (const float*)d_in[4];
  const float* Wk  = (const float*)d_in[5];
  const float* bk  = (const float*)d_in[6];
  const float* Wv  = (const float*)d_in[7];
  const float* bv  = (const float*)d_in[8];
  const float* Wo  = (const float*)d_in[9];
  const float* bo  = (const float*)d_in[10];
  float* out = (float*)d_out;

  char* ws = (char*)d_ws;
  size_t off = 0;
  auto alloc = [&](size_t elems) -> __bf16* {
    __bf16* p = (__bf16*)(ws + off);
    off += (elems * sizeof(__bf16) + 255) & ~(size_t)255;
    return p;
  };
  __bf16* Xbf   = alloc((size_t)MROWS * CC);
  __bf16* WshT  = alloc((size_t)SHR * CC);
  __bf16* WqT   = alloc((size_t)(HH * PRV) * CC);
  __bf16* WkT   = alloc((size_t)(HH * PRV) * CC);
  __bf16* WvT   = alloc((size_t)CC * CC);
  __bf16* WoT   = alloc((size_t)CC * CC);
  __bf16* sh_o  = alloc((size_t)MROWS * SHR);
  __bf16* q_o   = alloc((size_t)MROWS * (HH * PRV));
  __bf16* k_o   = alloc((size_t)MROWS * (HH * PRV));
  __bf16* v_o   = alloc((size_t)MROWS * CC);
  __bf16* Qb    = alloc((size_t)BB * HH * TT * HD);
  __bf16* Kb    = alloc((size_t)BB * HH * TT * HD);
  __bf16* Vtb   = alloc((size_t)BB * HH * HD * TT);
  __bf16* Yb    = alloc((size_t)MROWS * CC);
  (void)ws_size; (void)n_in; (void)in_sizes; (void)out_size;

  const int THR = 256;
  {
    int n = MROWS * CC;
    cvt_f32_bf16<<<(n + THR - 1) / THR, THR, 0, stream>>>(x, Xbf, n);
  }
  transpose_bf16<<<(CC * SHR + THR - 1) / THR, THR, 0, stream>>>(Wsh, WshT, CC, SHR);
  transpose_bf16<<<(CC * HH * PRV + THR - 1) / THR, THR, 0, stream>>>(Wq, WqT, CC, HH * PRV);
  transpose_bf16<<<(CC * HH * PRV + THR - 1) / THR, THR, 0, stream>>>(Wk, WkT, CC, HH * PRV);
  transpose_bf16<<<(CC * CC + THR - 1) / THR, THR, 0, stream>>>(Wv, WvT, CC, CC);
  transpose_bf16<<<(CC * CC + THR - 1) / THR, THR, 0, stream>>>(Wo, WoT, CC, CC);

  // waves = (M/64) * ceil((N/16)/4);  8 waves per block
  auto gemm_blocks = [](int M, int N) {
    int nt16 = N >> 4;
    int ngrp = (nt16 + 3) >> 2;
    int waves = (M >> 6) * ngrp;
    return (waves + 7) / 8;
  };
  gemm_wmma<0><<<gemm_blocks(MROWS, SHR),       THR, 0, stream>>>(Xbf, WshT, bsh, sh_o, MROWS, SHR,      CC);
  gemm_wmma<0><<<gemm_blocks(MROWS, HH * PRV),  THR, 0, stream>>>(Xbf, WqT,  bq,  q_o,  MROWS, HH * PRV, CC);
  gemm_wmma<0><<<gemm_blocks(MROWS, HH * PRV),  THR, 0, stream>>>(Xbf, WkT,  bk,  k_o,  MROWS, HH * PRV, CC);
  gemm_wmma<0><<<gemm_blocks(MROWS, CC),        THR, 0, stream>>>(Xbf, WvT,  bv,  v_o,  MROWS, CC,       CC);

  {
    int n = BB * HH * TT * HD;
    repack_qkv<<<(n + THR - 1) / THR, THR, 0, stream>>>(sh_o, q_o, k_o, v_o, Qb, Kb, Vtb);
  }

  flash_attn<<<(BB * HH * (TT / 16)) / 8, THR, 0, stream>>>(Qb, Kb, Vtb, Yb);

  gemm_wmma<1><<<gemm_blocks(MROWS, CC), THR, 0, stream>>>(Yb, WoT, bo, out, MROWS, CC, CC);
}
